// TensorBSplinesModel_51608327029238
// MI455X (gfx1250) — matrix operand order repaired and checked by reading, not compiled, not run on hardware
//
#include <hip/hip_runtime.h>

typedef __attribute__((ext_vector_type(16))) _Float16     v16h;
typedef __attribute__((ext_vector_type(8)))  float        v8f;
typedef __attribute__((ext_vector_type(8)))  unsigned int v8u;

#define INPUT_DIM  16
#define OUTPUT_DIM 16
#define BASIS_DIM  64
#define BATCH      131072
#define WAVES_PER_BLOCK 4
#define KFLAT (INPUT_DIM * BASIS_DIM)          /* 1024 */
#define WTILE_HALVES (OUTPUT_DIM * KFLAT)      /* 16384 f16 = 32KB */
#define BTILE_HALVES (16 * KFLAT)              /* 16384 f16 = 32KB per wave */

// Knots: t[0..2]=-0.1, t[3..64]=linspace(-0.1,1.1,62), t[65..67]=1.1
// => t[i] = -0.1 + h*clamp(i-3, 0, 61), h = 1.2/61
__device__ __forceinline__ float knotf(int i) {
    const float h = 1.2f / 61.0f;
    float c = (float)(i - 3);
    c = fminf(fmaxf(c, 0.0f), 61.0f);
    return -0.1f + h * c;
}

__global__ __launch_bounds__(32 * WAVES_PER_BLOCK)
void kan_bspline_wmma(const float* __restrict__ x,
                      const float* __restrict__ A,
                      const float* __restrict__ bias,
                      float* __restrict__ out) {
    extern __shared__ _Float16 smem[];   // [WTILE | per-wave basis tiles]
    const int tid   = threadIdx.x;
    const int lane  = tid & 31;
    const int wave  = tid >> 5;
    const int waveId = blockIdx.x * WAVES_PER_BLOCK + wave;
    const int b0    = waveId * 16;       // 16 batch rows per wave
    const int row   = lane & 15;
    const int group = lane >> 4;

    // ============ stage weights f32 -> f16 into LDS (block-wide) ============
    {
        const float4* Aq  = (const float4*)A;       // 4096 float4
        uint2*        Wh2 = (uint2*)smem;           // 4096 x 4 halves
        #pragma unroll
        for (int i = 0; i < 4096 / (32 * WAVES_PER_BLOCK); ++i) {
            int idx = tid + i * (32 * WAVES_PER_BLOCK);
            float4 w = Aq[idx];
            union { _Float16 h[2]; unsigned int u; } p0, p1;
            p0.h[0] = (_Float16)w.x; p0.h[1] = (_Float16)w.y;
            p1.h[0] = (_Float16)w.z; p1.h[1] = (_Float16)w.w;
            uint2 pr; pr.x = p0.u; pr.y = p1.u;
            Wh2[idx] = pr;                          // ds_store_b64
        }
    }
    __syncthreads();

    _Float16* bas = smem + WTILE_HALVES + wave * BTILE_HALVES;

    // ============ zero this wave's 16x1024 f16 basis tile ============
    {
        uint4* bz = (uint4*)bas;                    // 2048 x 16B
        uint4 z = make_uint4(0u, 0u, 0u, 0u);
        #pragma unroll
        for (int i = 0; i < 64; ++i)
            bz[lane + i * 32] = z;                  // ds_store_b128
    }

    // ==== per-lane cubic B-spline for 8 (row,dim) pairs; scatter 4 halves each ====
    {
        const float4* xw = (const float4*)(x + (size_t)b0 * INPUT_DIM);
        const float invh = 61.0f / 1.2f;
        #pragma unroll
        for (int i = 0; i < 2; ++i) {
            float4 xq = xw[lane * 2 + i];
            float xa[4] = {xq.x, xq.y, xq.z, xq.w};
            #pragma unroll
            for (int q = 0; q < 4; ++q) {
                int f  = lane * 8 + i * 4 + q;      // flat index into 16x16 x-tile
                int r_ = f >> 4;                    // batch row within tile
                int d_ = f & 15;                    // input dim
                float u = xa[q];
                int j = 3 + (int)floorf((u + 0.1f) * invh);
                j = max(3, min(j, 63));
                // left[k] = u - t[j+1-k], right[k] = t[j+k] - u
                float l1 = u - knotf(j);
                float r1 = knotf(j + 1) - u;
                float l2 = u - knotf(j - 1);
                float r2 = knotf(j + 2) - u;
                float l3 = u - knotf(j - 2);
                float r3 = knotf(j + 3) - u;
                float N0 = 1.0f, N1, N2, N3, sv, tm;
                tm = N0 * __builtin_amdgcn_rcpf(r1 + l1);
                N0 = r1 * tm; N1 = l1 * tm;
                sv = 0.0f;
                tm = N0 * __builtin_amdgcn_rcpf(r1 + l2); N0 = sv + r1 * tm; sv = l2 * tm;
                tm = N1 * __builtin_amdgcn_rcpf(r2 + l1); N1 = sv + r2 * tm; sv = l1 * tm;
                N2 = sv;
                sv = 0.0f;
                tm = N0 * __builtin_amdgcn_rcpf(r1 + l3); N0 = sv + r1 * tm; sv = l3 * tm;
                tm = N1 * __builtin_amdgcn_rcpf(r2 + l2); N1 = sv + r2 * tm; sv = l2 * tm;
                tm = N2 * __builtin_amdgcn_rcpf(r3 + l1); N2 = sv + r3 * tm; sv = l1 * tm;
                N3 = sv;
                _Float16* dst = bas + r_ * KFLAT + d_ * BASIS_DIM + (j - 3);
                dst[0] = (_Float16)N0;              // ds_store_b16 x4 (2B aligned)
                dst[1] = (_Float16)N1;
                dst[2] = (_Float16)N2;
                dst[3] = (_Float16)N3;
            }
        }
    }
    __syncthreads();   // defensive; uniform across all waves

    // ============ K-loop: 32 chunks of K=32, one WMMA each ============
    v8f c = {};
    const int   o    = lane & 15;
    const uint4* basq = (const uint4*)bas;          // 8 halves per uint4
    const uint4* wq   = (const uint4*)smem;
    #pragma unroll
    for (int kc = 0; kc < 32; ++kc) {
        int aoff = (row * KFLAT + kc * 32 + group * 8) >> 3;
        uint4 a0 = basq[aoff];                      // halves 0..7
        uint4 a1 = basq[aoff + 2];                  // halves 8..15 (K +16)
        int boff = (o * KFLAT + kc * 32 + group * 8) >> 3;
        uint4 q0 = wq[boff];
        uint4 q1 = wq[boff + 2];
        v8u av = {a0.x, a0.y, a0.z, a0.w, a1.x, a1.y, a1.z, a1.w};
        v8u bv = {q0.x, q0.y, q0.z, q0.w, q1.x, q1.y, q1.z, q1.w};
        v16h af = __builtin_bit_cast(v16h, av);
        v16h bf = __builtin_bit_cast(v16h, bv);
        c = __builtin_amdgcn_wmma_f32_16x16x32_f16(
                /*neg_a=*/false, af, /*neg_b=*/false, bf,
                /*c_mod=*/(short)0, c, /*reuse_a=*/false, /*reuse_b=*/false);
    }

    // ===== bias + store: VGPR r, lane L -> row group*8+r, col lane%16 =====
    float bb = bias[o];
    float* orow = out + (size_t)(b0 + group * 8) * OUTPUT_DIM + o;
    #pragma unroll
    for (int r = 0; r < 8; ++r)
        orow[r * OUTPUT_DIM] = c[r] + bb;
}

extern "C" void kernel_launch(void* const* d_in, const int* in_sizes, int n_in,
                              void* d_out, int out_size, void* d_ws, size_t ws_size,
                              hipStream_t stream) {
    (void)in_sizes; (void)n_in; (void)d_ws; (void)ws_size; (void)out_size;
    const float* x    = (const float*)d_in[0];   // [131072, 16]
    const float* A    = (const float*)d_in[1];   // [16, 16, 64]
    const float* bias = (const float*)d_in[2];   // [16]
    float* out        = (float*)d_out;           // [131072, 16]

    const int threads = 32 * WAVES_PER_BLOCK;                     // 128
    const int blocks  = (BATCH / 16) / WAVES_PER_BLOCK;           // 2048
    const size_t shmem = (size_t)(WTILE_HALVES + WAVES_PER_BLOCK * BTILE_HALVES)
                         * sizeof(_Float16);                      // 160 KB
    kan_bspline_wmma<<<blocks, threads, shmem, stream>>>(x, A, bias, out);
}
